// Attention_52037823759037
// MI455X (gfx1250) — matrix operand order repaired and checked by reading, not compiled, not run on hardware
//
#include <hip/hip_runtime.h>
#include <math.h>

typedef __attribute__((ext_vector_type(16))) __bf16 v16bf;
typedef __attribute__((ext_vector_type(8)))  __bf16 v8bf;
typedef __attribute__((ext_vector_type(4)))  __bf16 v4bf;
typedef __attribute__((ext_vector_type(8)))  float  v8f;
typedef __attribute__((ext_vector_type(4)))  float  v4f;

static constexpr int Bb = 2, Tt = 2048, Cc = 1024, Hh = 16, HD = 64;
static constexpr int BH = Bb * Hh;

__device__ __forceinline__ v8bf ld8(const __bf16* p) { return *(const v8bf*)p; }

__device__ __forceinline__ v16bf cat16(v8bf lo, v8bf hi) {
  v16bf r;
#pragma unroll
  for (int i = 0; i < 8; ++i) { r[i] = lo[i]; r[i + 8] = hi[i]; }
  return r;
}

__device__ __forceinline__ v8f wmma_bf16(v16bf a, v16bf b, v8f c) {
  // D(f32,16x16) = A(bf16,16x32) * B(bf16,32x16) + C
  return __builtin_amdgcn_wmma_f32_16x16x32_bf16(false, a, false, b, (short)0, c,
                                                 false, false);
}

// CDNA5 async global->LDS copy, 16 bytes per lane (GLOBAL_LOAD_ASYNC_TO_LDS_B128,
// tracked by ASYNCcnt). LDS offset = low 32 bits of the generic pointer (ISA 10.2).
__device__ __forceinline__ void async_ld16(void* lds_dst, const void* gsrc) {
  unsigned loff = (unsigned)(size_t)lds_dst;
  asm volatile("global_load_async_to_lds_b128 %0, %1, off"
               :: "v"(loff), "v"(gsrc)
               : "memory");
}
__device__ __forceinline__ void wait_async0() {
  asm volatile("s_wait_asynccnt 0x0" ::: "memory");
}

// ---------------------------------------------------------------------------
// Prep kernels (memory-bound, ~1us total at 23.3 TB/s): fp32 -> bf16, and
// weight transpose+convert so GEMM tiles can be pure async B128 copies.
// ---------------------------------------------------------------------------
__global__ __launch_bounds__(256)
void cvt_bf16(const float* __restrict__ X, __bf16* __restrict__ Xb) {
  const size_t i = (size_t)blockIdx.x * 256 + threadIdx.x;
  v4f v = *(const v4f*)(X + i * 4);
  v4bf o;
#pragma unroll
  for (int k = 0; k < 4; ++k) o[k] = (__bf16)v[k];
  *(v4bf*)(Xb + i * 4) = o;
}

// W fp32 [K][N]  ->  Wt bf16 [N][K]
__global__ __launch_bounds__(256)
void transpose_cvt(const float* __restrict__ W, __bf16* __restrict__ Wt,
                   int K, int N) {
  __shared__ float t[32][33];
  const int n0 = blockIdx.x * 32, k0 = blockIdx.y * 32;
  const int c = threadIdx.x & 31, r8 = threadIdx.x >> 5;
#pragma unroll
  for (int i = 0; i < 4; ++i) {
    const int r = r8 + i * 8;
    t[r][c] = W[(size_t)(k0 + r) * N + n0 + c];
  }
  __syncthreads();
#pragma unroll
  for (int i = 0; i < 4; ++i) {
    const int r = r8 + i * 8;
    Wt[(size_t)(n0 + r) * K + k0 + c] = (__bf16)t[c][r];
  }
}

// ---------------------------------------------------------------------------
// bf16 WMMA GEMM: C = A[MxK] * Bt[NxK]^T (+bias), block 256x128, 8 waves in a
// 4x2 grid, each wave a 64x64 sub-tile = 4x4 WMMA accumulators (16 WMMAs per
// 8 fragment-load pairs -> 1:1 ds:wmma). BK=32, double-buffered async tiles.
//   MODE 0: fp32 output (+bias)      MODE 1: scatter Q/K/Vt bf16 (V transposed)
// ---------------------------------------------------------------------------
template <int MODE>
__global__ __launch_bounds__(256)
void gemm256(const __bf16* __restrict__ A, const __bf16* __restrict__ Bt,
             const float* __restrict__ bias, int M, int N, int K,
             float* __restrict__ Cout,
             __bf16* __restrict__ Qo, __bf16* __restrict__ Ko, __bf16* __restrict__ Vo) {
  __shared__ __align__(16) __bf16 As[2][256][40];   // [m][k], 80B rows (16B mult)
  __shared__ __align__(16) __bf16 Bts[2][128][40];  // [n][k]

  const int tid = threadIdx.x;
  const int lane = tid & 31, wid = tid >> 5;
  const int half = lane >> 4, l16 = lane & 15;
  const int wm = wid & 3, wn = wid >> 2;  // 4x2 wave grid
  const int m0 = blockIdx.y * 256;
  const int n0 = blockIdx.x * 128;

  v8f acc[4][4] = {};

  // one 256x32 A tile (1024 chunks) + 128x32 Bt tile (512 chunks), 6/thread
  auto issue = [&](int buf, int k0) {
#pragma unroll
    for (int i = 0; i < 4; ++i) {
      const int e = tid + i * 256;
      const int r = e >> 2, seg = (e & 3) * 8;
      async_ld16(&As[buf][r][seg], A + (size_t)(m0 + r) * K + k0 + seg);
    }
#pragma unroll
    for (int i = 0; i < 2; ++i) {
      const int e = tid + i * 256;
      const int r = e >> 2, seg = (e & 3) * 8;
      async_ld16(&Bts[buf][r][seg], Bt + (size_t)(n0 + r) * K + k0 + seg);
    }
  };

  const int nk = K / 32;
  issue(0, 0);
  for (int it = 0; it < nk; ++it) {
    const int cur = it & 1;
    wait_async0();
    __syncthreads();
    if (it + 1 < nk) issue(1 - cur, (it + 1) * 32);  // overlap with WMMA below

    // A fragments: row per lane; K = {half*8+0..7, half*8+16..23}
    v16bf afr[4];
#pragma unroll
    for (int tm = 0; tm < 4; ++tm) {
      const __bf16* ap = &As[cur][wm * 64 + tm * 16 + l16][half * 8];
      afr[tm] = cat16(ld8(ap), ld8(ap + 16));
    }
    // B fragments: col per lane; K = {half*8+0..7, half*8+16..23}
#pragma unroll
    for (int tn = 0; tn < 4; ++tn) {
      const __bf16* bp = &Bts[cur][wn * 64 + tn * 16 + l16][half * 8];
      v16bf bfr = cat16(ld8(bp), ld8(bp + 16));
#pragma unroll
      for (int tm = 0; tm < 4; ++tm)
        acc[tm][tn] = wmma_bf16(afr[tm], bfr, acc[tm][tn]);
    }
  }

  // Epilogue. C layout: VGPR j -> row j + 8*half, col = l16.
  // Within a lane, j walks the token/row dimension (stride 1 rows).
#pragma unroll
  for (int tm = 0; tm < 4; ++tm) {
    const int row0 = m0 + wm * 64 + tm * 16 + half * 8;  // 8-aligned
#pragma unroll
    for (int tn = 0; tn < 4; ++tn) {
      const int col = n0 + wn * 64 + tn * 16 + l16;
      const float bv = bias ? bias[col] : 0.f;
      if (MODE == 0) {
        float* dst = Cout + (size_t)row0 * N + col;
#pragma unroll
        for (int j = 0; j < 8; ++j) { *dst = acc[tm][tn][j] + bv; dst += N; }
      } else {
        const int which = col >> 10;
        const int c = col & 1023;
        const int h = c >> 6, hd = c & 63;
        const int b = row0 >> 11, t0 = row0 & 2047;  // 8-row run stays in one b
        const size_t bh = (size_t)(b * Hh + h);
        if (which == 2) {
          // V^T [bh][hd][t]: j walks t -> one contiguous 16B store
          v8bf pk;
#pragma unroll
          for (int j = 0; j < 8; ++j) pk[j] = (__bf16)(acc[tm][tn][j] + bv);
          *(v8bf*)(Vo + (bh * HD + hd) * Tt + t0) = pk;
        } else {
          __bf16* dst = (which ? Ko : Qo) + (bh * Tt + t0) * HD + hd;
#pragma unroll
          for (int j = 0; j < 8; ++j) {
            *dst = (__bf16)(acc[tm][tn][j] + bv);
            dst += HD;
          }
        }
      }
    }
  }
}

// ---------------------------------------------------------------------------
// Causal flash attention. Block = (head, 64 query rows), 4 waves x 16 rows.
// K tile [key][hd] and pre-transposed V tile [hd][key] stream through LDS via
// double-buffered async copies. Softmax in raw-score units, 1/sqrt(d) folded
// into the exp2 base; masking only on the diagonal block. Row-max reduction is
// per block (alpha must be row-uniform); the row-SUM recurrence is linear, so
// each lane carries a per-lane partial and the 16-lane reduction happens once.
// ---------------------------------------------------------------------------
__global__ __launch_bounds__(128)
void flash_attn(const __bf16* __restrict__ Q, const __bf16* __restrict__ Kk,
                const __bf16* __restrict__ Vt, __bf16* __restrict__ Y) {
  __shared__ __align__(16) __bf16 Ks[2][64][72];   // [key][hd], 144B rows
  __shared__ __align__(16) __bf16 Vts[2][64][72];  // [hd][key]
  __shared__ __align__(16) __bf16 Ps[4][16][72];   // per-wave P [row][key]

  const int tid = threadIdx.x;
  const int lane = tid & 31, wid = tid >> 5;
  const int half = lane >> 4, l16 = lane & 15;
  const int qblk = blockIdx.x;
  const int bh = blockIdx.y;
  const int b = bh >> 4, h = bh & 15;
  const int qrow0 = qblk * 64 + wid * 16;

  // exp2 base with 1/sqrt(64)*log2(e) folded in
  const float C2 = 0.125f * 1.44269504f;

  // Q A-fragments, register-resident for the whole kernel (two K=32 chunks)
  const __bf16* Qp = Q + ((size_t)bh * Tt + qrow0 + l16) * HD;
  v16bf aq[2];
#pragma unroll
  for (int c = 0; c < 2; ++c) {
    const __bf16* p = Qp + c * 32 + half * 8;
    aq[c] = cat16(ld8(p), ld8(p + 16));
  }

  const __bf16* Kbase = Kk + (size_t)bh * Tt * HD;
  const __bf16* Vbase = Vt + (size_t)bh * HD * Tt;

  // issue rows 0..63, cols 0..31 of the K and Vt tiles: 512 x 16B chunks, 4 per thread
  auto issue = [&](int buf, int key0) {
#pragma unroll
    for (int i = 0; i < 2; ++i) {
      const int e = tid + i * 128;
      const int r = e >> 2, seg = (e & 3) * 8;
      async_ld16(&Ks[buf][r][seg],  Kbase + (size_t)(key0 + r) * HD + seg);
      async_ld16(&Vts[buf][r][seg], Vbase + (size_t)r * Tt + key0 + seg);
    }
  };

  v8f o[4] = {};
  float m[8], lp[8];
#pragma unroll
  for (int j = 0; j < 8; ++j) { m[j] = -1e30f; lp[j] = 0.f; }

  auto process = [&](int cur, int key0, bool diag) {
    // S = Q K^T : 4 key sub-tiles, K-dim = HD = 64 (2 chained WMMAs)
    v8f s[4];
#pragma unroll
    for (int tn = 0; tn < 4; ++tn) {
      const __bf16* bp0 = &Ks[cur][tn * 16 + l16][half * 8];
      const __bf16* bp1 = &Ks[cur][tn * 16 + l16][32 + half * 8];
      v16bf b0 = cat16(ld8(bp0), ld8(bp0 + 16));
      v16bf b1 = cat16(ld8(bp1), ld8(bp1 + 16));
      v8f z = {};
      z = wmma_bf16(aq[0], b0, z);
      s[tn] = wmma_bf16(aq[1], b1, z);
    }

    if (diag) {  // causal mask, only the diagonal key block needs it
#pragma unroll
      for (int tn = 0; tn < 4; ++tn) {
#pragma unroll
        for (int j = 0; j < 8; ++j) {
          const int row = qrow0 + j + half * 8;
          const int col = key0 + tn * 16 + l16;
          if (col > row) s[tn][j] = -1e30f;
        }
      }
    }

    // row max: 3x max3 + 16-lane butterfly (stays inside the half)
    float mt[8];
#pragma unroll
    for (int j = 0; j < 8; ++j) {
      float v = fmaxf(fmaxf(s[0][j], s[1][j]), fmaxf(s[2][j], s[3][j]));
#pragma unroll
      for (int d = 1; d < 16; d <<= 1) v = fmaxf(v, __shfl_xor(v, d, 32));
      mt[j] = v;
    }
    float alpha[8], mn[8];
#pragma unroll
    for (int j = 0; j < 8; ++j) {
      mn[j] = fmaxf(m[j], mt[j]);
      alpha[j] = exp2f((m[j] - mn[j]) * C2);
      m[j] = mn[j];
    }
    // p = exp2((s-m)*C2); per-lane partial sum (cross-lane reduce deferred)
#pragma unroll
    for (int j = 0; j < 8; ++j) lp[j] *= alpha[j];
#pragma unroll
    for (int tn = 0; tn < 4; ++tn) {
#pragma unroll
      for (int j = 0; j < 8; ++j) {
        const float p = exp2f((s[tn][j] - mn[j]) * C2);
        lp[j] += p;
        Ps[wid][j + half * 8][tn * 16 + l16] = (__bf16)p;  // per-wave region
      }
    }
#pragma unroll
    for (int t = 0; t < 4; ++t)
#pragma unroll
      for (int j = 0; j < 8; ++j) o[t][j] *= alpha[j];

    // P @ V (P reread from LDS in A-layout; same-wave LDS stays in order)
    v16bf ap[2];
#pragma unroll
    for (int c = 0; c < 2; ++c) {
      const __bf16* p = &Ps[wid][l16][c * 32 + half * 8];
      ap[c] = cat16(ld8(p), ld8(p + 16));
    }
#pragma unroll
    for (int t = 0; t < 4; ++t) {
      const __bf16* vp0 = &Vts[cur][t * 16 + l16][half * 8];
      const __bf16* vp1 = &Vts[cur][t * 16 + l16][32 + half * 8];
      v16bf b0 = cat16(ld8(vp0), ld8(vp0 + 16));
      v16bf b1 = cat16(ld8(vp1), ld8(vp1 + 16));
      o[t] = wmma_bf16(ap[0], b0, o[t]);
      o[t] = wmma_bf16(ap[1], b1, o[t]);
    }
  };

  issue(0, 0);
  int kb = 0;
  for (; kb < qblk; ++kb) {  // off-diagonal blocks: no masking
    const int cur = kb & 1;
    wait_async0();
    __syncthreads();
    issue(1 - cur, (kb + 1) * 64);  // overlap next tile with WMMA
    process(cur, kb * 64, false);
  }
  {  // diagonal block
    const int cur = kb & 1;
    wait_async0();
    __syncthreads();
    process(cur, kb * 64, true);
  }

  // one deferred 16-lane reduction of the row-sum partials
  float linv[8];
#pragma unroll
  for (int j = 0; j < 8; ++j) {
    float v = lp[j];
#pragma unroll
    for (int d = 1; d < 16; d <<= 1) v += __shfl_xor(v, d, 32);
    linv[j] = 1.f / v;
  }

  // normalize, stage wave's 16x64 tile through (now free) Ps, then 4 coalesced
  // 16B global stores per lane into Y[B,T,C]
#pragma unroll
  for (int t = 0; t < 4; ++t)
#pragma unroll
    for (int j = 0; j < 8; ++j)
      Ps[wid][j + half * 8][t * 16 + l16] = (__bf16)(o[t][j] * linv[j]);

  __bf16* Ybase = Y + ((size_t)b * Tt + qrow0) * Cc + h * HD;
#pragma unroll
  for (int i = 0; i < 4; ++i) {
    const int cid = lane + 32 * i;          // 128 chunks: 16 rows x 8
    const int r = cid >> 3, seg = (cid & 7) * 8;
    *(v8bf*)(Ybase + (size_t)r * Cc + seg) = ld8(&Ps[wid][r][seg]);
  }
}

// ---------------------------------------------------------------------------
extern "C" void kernel_launch(void* const* d_in, const int* in_sizes, int n_in,
                              void* d_out, int out_size, void* d_ws, size_t ws_size,
                              hipStream_t stream) {
  (void)in_sizes; (void)n_in; (void)out_size; (void)ws_size;
  const float* x     = (const float*)d_in[0];
  const float* w_qkv = (const float*)d_in[1];
  const float* b_qkv = (const float*)d_in[2];
  const float* w_out = (const float*)d_in[3];
  const float* b_out = (const float*)d_in[4];
  float* out = (float*)d_out;

  const size_t nTok = (size_t)Bb * Tt;          // 4096
  const size_t nQ   = (size_t)BH * Tt * HD;     // 4M elems
  __bf16* Xb     = (__bf16*)d_ws;               // 4M
  __bf16* WqkvT  = Xb + nTok * Cc;              // 3M   [3C][C]
  __bf16* WoutT  = WqkvT + (size_t)3 * Cc * Cc; // 1M   [C][C]
  __bf16* Qb     = WoutT + (size_t)Cc * Cc;     // 4M   [B,H,T,HD]
  __bf16* Kb     = Qb + nQ;                     // 4M   [B,H,T,HD]
  __bf16* Vtb    = Kb + nQ;                     // 4M   [B,H,HD,T] (transposed)
  __bf16* Yb     = Vtb + nQ;                    // 4M   [B,T,C]    (~48 MB total)

  const int M = (int)nTok;

  // 0) fp32 -> bf16 conversions / weight transposes (tiny, bandwidth-bound)
  cvt_bf16<<<(unsigned)(nTok * Cc / 1024), 256, 0, stream>>>(x, Xb);
  transpose_cvt<<<dim3(3 * Cc / 32, Cc / 32), 256, 0, stream>>>(w_qkv, WqkvT, Cc, 3 * Cc);
  transpose_cvt<<<dim3(Cc / 32, Cc / 32), 256, 0, stream>>>(w_out, WoutT, Cc, Cc);

  // 1) QKV projection (M=4096, N=3072, K=1024) -> bf16 Q/K/Vt
  dim3 g1(3 * Cc / 128, M / 256);
  gemm256<1><<<g1, 256, 0, stream>>>(Xb, WqkvT, b_qkv, M, 3 * Cc, Cc,
                                     nullptr, Qb, Kb, Vtb);

  // 2) causal flash attention -> bf16 Y
  dim3 g2(Tt / 64, BH);
  flash_attn<<<g2, 128, 0, stream>>>(Qb, Kb, Vtb, Yb);

  // 3) output projection (M=4096, N=1024, K=1024) -> fp32 out + bias
  dim3 g3(Cc / 128, M / 256);
  gemm256<0><<<g3, 256, 0, stream>>>(Yb, WoutT, b_out, M, Cc, Cc,
                                     out, nullptr, nullptr, nullptr);
}
